// TimeAwareCost_57621281243482
// MI455X (gfx1250) — compile-verified, matrix operand order checked by reading, not run on hardware
//
#include <hip/hip_runtime.h>
#include <hip/hip_bf16.h>
#include <math.h>

typedef __attribute__((ext_vector_type(16))) _Float16 v16h;
typedef __attribute__((ext_vector_type(8)))  _Float16 v8h;
typedef __attribute__((ext_vector_type(8)))  float    v8f;

namespace {
constexpr int N_  = 2048;   // rows of x / z
constexpr int K_  = 512;    // x feature dim
constexpr int DZ_ = 256;    // z feature dim
constexpr int NT_ = N_ / 16; // 128 WMMA tiles per dimension
}

union ABFrag { v16h v; v8h h[2]; };

// ---------------------------------------------------------------------------
// scal layout (floats): [0]=Dmax bits  [1]=R_last  [2]=latent_sum
//                       [3]=d_acf      [4]=mismatch flag (int bits)
// ---------------------------------------------------------------------------
__global__ void init_scalars_kernel(float* scal) {
    int t = threadIdx.x;
    if (t < 8) scal[t] = 0.0f;
}

// One block per row: norm, inverse norm (with +1e-8), squared norm of the
// normalized row (matches sum(xn*xn) in the reference).
__global__ __launch_bounds__(256)
void row_norm_kernel(const float* __restrict__ X, float* __restrict__ invn,
                     float* __restrict__ nx2) {
    const int row = blockIdx.x;
    const int tid = threadIdx.x;
    const float* p = X + (size_t)row * K_;
    float s = 0.0f;
    for (int c = tid; c < K_; c += 256) { float v = p[c]; s += v * v; }
    __shared__ float red[256];
    red[tid] = s; __syncthreads();
    for (int off = 128; off > 0; off >>= 1) {
        if (tid < off) red[tid] += red[tid + off];
        __syncthreads();
    }
    if (tid == 0) {
        float nrm = sqrtf(red[0]);
        float iv  = 1.0f / (nrm + 1e-8f);
        invn[row] = iv;
        nx2[row]  = (nrm * iv) * (nrm * iv);
    }
}

// Normalize rows and pack to f16 (values are in [-1,1] -> safe for f16).
__global__ void normalize_f16_kernel(const float* __restrict__ X,
                                     const float* __restrict__ invn,
                                     _Float16* __restrict__ Xh) {
    int idx    = blockIdx.x * blockDim.x + threadIdx.x;
    int stride = gridDim.x * blockDim.x;
    const int n = N_ * K_;
    for (int i = idx; i < n; i += stride) {
        Xh[i] = (_Float16)(X[i] * invn[i / K_]);
    }
}

// One wave32 per 16x16 tile of D. G = Xn * Yn^T via v_wmma_f32_16x16x32_f16,
// epilogue: D = sqrt(max(nx2_i + ny2_j - 2G, 0)), track global max of D.
__global__ __launch_bounds__(32)
void wmma_dist_kernel(const _Float16* __restrict__ Xh,
                      const _Float16* __restrict__ Yh,
                      const float* __restrict__ nx2i,
                      const float* __restrict__ nx2j,
                      float* __restrict__ D,
                      unsigned int* __restrict__ dmax_bits) {
    const int tile = blockIdx.x;
    const int tm = tile / NT_;          // tile row (i block)
    const int tn = tile % NT_;          // tile col (j block)
    const int lane = threadIdx.x;

    // 16-bit A-fragment layout (ISA 7.12.2): lanes 0-15 hold K 0..7 / 16..23
    // of row M=lane; lanes 16-31 hold K 8..15 / 24..31 of row M=lane-16.
    // B fragment for X*Y^T loads identically with lane -> column (= Y row).
    const int r    = lane & 15;
    const int ksel = (lane >> 4) * 8;
    const _Float16* arow = Xh + (size_t)(tm * 16 + r) * K_ + ksel;
    const _Float16* brow = Yh + (size_t)(tn * 16 + r) * K_ + ksel;

    v8f acc = {};
    for (int k0 = 0; k0 < K_; k0 += 32) {
        __builtin_prefetch(arow + k0 + 64, 0, 1);
        __builtin_prefetch(brow + k0 + 64, 0, 1);
        ABFrag a, b;
        a.h[0] = *(const v8h*)(arow + k0);
        a.h[1] = *(const v8h*)(arow + k0 + 16);
        b.h[0] = *(const v8h*)(brow + k0);
        b.h[1] = *(const v8h*)(brow + k0 + 16);
        acc = __builtin_amdgcn_wmma_f32_16x16x32_f16(
            /*neg_a=*/false, a.v, /*neg_b=*/false, b.v,
            /*c_mod=*/(short)0, acc, /*reuse_a=*/false, /*reuse_b=*/false);
    }

    // C/D layout: lanes 0-15: N=lane, VGPR v -> M=v; lanes 16-31: N=lane-16,
    // VGPR v -> M=v+8.
    const int col     = lane & 15;
    const int rowbase = (lane >> 4) * 8;
    float lmax = 0.0f;
    #pragma unroll
    for (int v = 0; v < 8; ++v) {
        int gi = tm * 16 + rowbase + v;
        int gj = tn * 16 + col;
        float sq = nx2i[gi] + nx2j[gj] - 2.0f * acc[v];
        float d  = sqrtf(fmaxf(sq, 0.0f));
        D[(size_t)gi * N_ + gj] = d;
        lmax = fmaxf(lmax, d);
    }
    // wave max reduction, then one atomic per wave (d >= 0 so float bits are
    // monotone under unsigned compare)
    for (int off = 16; off > 0; off >>= 1)
        lmax = fmaxf(lmax, __shfl_xor(lmax, off, 32));
    if (lane == 0) atomicMax(dmax_bits, __float_as_uint(lmax));
}

// Anti-diagonal DTW wavefront on UNNORMALIZED D (linear in scale; finalize
// divides by Dmax+1e-8). One workgroup of 1024 threads (32 waves) on one WGP,
// 2 cells per thread per diagonal, 3 rotating diagonals in LDS.
__global__ __launch_bounds__(1024)
void dtw_kernel(const float* __restrict__ D, float* __restrict__ Rout) {
    __shared__ float buf[3][N_];
    const float INF = __builtin_huge_valf();
    const int tid = threadIdx.x;

    buf[0][tid] = INF; buf[0][tid + 1024] = INF;
    buf[1][tid] = INF; buf[1][tid + 1024] = INF;
    __syncthreads();

    int p2 = 0, p1 = 1, cu = 2;
    for (int k = 0; k < 2 * N_ - 1; ++k) {
        #pragma unroll
        for (int h = 0; h < 2; ++h) {
            const int i = tid + h * 1024;
            const int j = k - i;
            float cur = INF;
            if (j >= 0 && j < N_) {
                float d    = D[(size_t)i * N_ + j];
                float up   = (i > 0) ? buf[p1][i - 1] : INF;           // R[i-1,j]
                float left = buf[p1][i];                               // R[i,j-1]
                float diag = (i > 0) ? buf[p2][i - 1]
                                     : ((k == 0) ? 0.0f : INF);        // R[i-1,j-1]
                cur = d + fminf(fminf(up, left), diag);
            }
            buf[cu][i] = cur;
        }
        __syncthreads();
        int t = p2; p2 = p1; p1 = cu; cu = t;
    }
    if (tid == 0) Rout[0] = buf[p1][N_ - 1];  // R at (n-1, m-1), unscaled
}

// sum (zi - zj)^2 -> scal[2]
__global__ __launch_bounds__(256)
void latent_kernel(const float* __restrict__ a, const float* __restrict__ b,
                   int n, float* __restrict__ acc) {
    int idx    = blockIdx.x * blockDim.x + threadIdx.x;
    int stride = gridDim.x * blockDim.x;
    float s = 0.0f;
    for (int i = idx; i < n; i += stride) { float d = a[i] - b[i]; s += d * d; }
    __shared__ float red[256];
    red[threadIdx.x] = s; __syncthreads();
    for (int off = 128; off > 0; off >>= 1) {
        if (threadIdx.x < off) red[threadIdx.x] += red[threadIdx.x + off];
        __syncthreads();
    }
    if (threadIdx.x == 0) atomicAdd(acc, red[0]);
}

__device__ void column_acf(const float* __restrict__ Z, int d, float* a) {
    float mean = 0.0f;
    for (int t = 0; t < N_; ++t) mean += Z[(size_t)t * DZ_ + d];
    mean *= (1.0f / N_);
    float s[5];
    #pragma unroll
    for (int lag = 1; lag <= 5; ++lag) {
        float sum = 0.0f;
        for (int t = 0; t < N_ - lag; ++t) {
            float z0 = Z[(size_t)t * DZ_ + d] - mean;
            float z1 = Z[(size_t)(t + lag) * DZ_ + d] - mean;
            sum += z0 * z1;
        }
        s[lag - 1] = sum;
    }
    float m = 0.0f;
    #pragma unroll
    for (int l = 0; l < 5; ++l) m = fmaxf(m, fabsf(s[l]));
    #pragma unroll
    for (int l = 0; l < 5; ++l) a[l] = s[l] / m;   // reference has no epsilon
}

// one block, 256 threads, one column per thread; writes scal[3]
__global__ __launch_bounds__(256)
void acf_diff_kernel(const float* __restrict__ Zi, const float* __restrict__ Zj,
                     float* __restrict__ out) {
    const int d = threadIdx.x;
    float ai[5], aj[5];
    column_acf(Zi, d, ai);
    column_acf(Zj, d, aj);
    float s = 0.0f;
    #pragma unroll
    for (int l = 0; l < 5; ++l) s += fabsf(ai[l] - aj[l]);
    __shared__ float red[256];
    red[d] = s; __syncthreads();
    for (int off = 128; off > 0; off >>= 1) {
        if (d < off) red[d] += red[d + off];
        __syncthreads();
    }
    if (d == 0) out[0] = red[0];
}

__global__ __launch_bounds__(256)
void mismatch_kernel(const float* __restrict__ a, const float* __restrict__ b,
                     int n, int* __restrict__ flag) {
    int idx    = blockIdx.x * blockDim.x + threadIdx.x;
    int stride = gridDim.x * blockDim.x;
    int local  = 0;
    for (int i = idx; i < n; i += stride)
        if (a[i] != b[i]) { local = 1; break; }
    if (local) atomicOr(flag, 1);
}

__global__ void finalize_kernel(const float* __restrict__ scal,
                                float* __restrict__ out) {
    float dmax     = __uint_as_float(__float_as_uint(scal[0]));
    float d_time   = fabsf(scal[1]) / (dmax + 1e-8f);
    float d_latent = sqrtf(scal[2]) / (float)N_;
    float d_acf    = scal[3];
    int   mismatch = __float_as_int(scal[4]);
    float cost = 0.4f * d_time + 0.4f * d_latent + 0.2f * d_acf;
    out[0] = mismatch ? cost : 0.0f;
}

// ---------------------------------------------------------------------------
extern "C" void kernel_launch(void* const* d_in, const int* in_sizes, int n_in,
                              void* d_out, int out_size, void* d_ws, size_t ws_size,
                              hipStream_t stream) {
    const float* xi = (const float*)d_in[0];   // [2048, 512]
    const float* xj = (const float*)d_in[1];   // [2048, 512]
    const float* zi = (const float*)d_in[2];   // [2048, 256]
    const float* zj = (const float*)d_in[3];   // [2048, 256]
    float* out = (float*)d_out;

    // workspace layout (~20.1 MB)
    char* ws = (char*)d_ws;
    size_t off = 0;
    float*    Dm  = (float*)(ws + off);    off += (size_t)N_ * N_ * sizeof(float);     // 16 MB
    _Float16* xh  = (_Float16*)(ws + off); off += (size_t)N_ * K_ * sizeof(_Float16);  // 2 MB
    _Float16* yh  = (_Float16*)(ws + off); off += (size_t)N_ * K_ * sizeof(_Float16);  // 2 MB
    float* invn_i = (float*)(ws + off);    off += N_ * sizeof(float);
    float* nx2_i  = (float*)(ws + off);    off += N_ * sizeof(float);
    float* invn_j = (float*)(ws + off);    off += N_ * sizeof(float);
    float* nx2_j  = (float*)(ws + off);    off += N_ * sizeof(float);
    float* scal   = (float*)(ws + off);    off += 64;

    init_scalars_kernel<<<1, 32, 0, stream>>>(scal);

    row_norm_kernel<<<N_, 256, 0, stream>>>(xi, invn_i, nx2_i);
    row_norm_kernel<<<N_, 256, 0, stream>>>(xj, invn_j, nx2_j);
    normalize_f16_kernel<<<512, 256, 0, stream>>>(xi, invn_i, xh);
    normalize_f16_kernel<<<512, 256, 0, stream>>>(xj, invn_j, yh);

    wmma_dist_kernel<<<NT_ * NT_, 32, 0, stream>>>(
        xh, yh, nx2_i, nx2_j, Dm, (unsigned int*)&scal[0]);

    dtw_kernel<<<1, 1024, 0, stream>>>(Dm, &scal[1]);

    latent_kernel<<<256, 256, 0, stream>>>(zi, zj, N_ * DZ_, &scal[2]);
    acf_diff_kernel<<<1, DZ_, 0, stream>>>(zi, zj, &scal[3]);
    mismatch_kernel<<<256, 256, 0, stream>>>(xi, xj, N_ * K_, (int*)&scal[4]);
    mismatch_kernel<<<256, 256, 0, stream>>>(zi, zj, N_ * DZ_, (int*)&scal[4]);

    finalize_kernel<<<1, 1, 0, stream>>>(scal, out);
}
